// Head_47631187312707
// MI455X (gfx1250) — matrix operand order repaired and checked by reading, not compiled, not run on hardware
//
#include <hip/hip_runtime.h>
#include <hip/hip_bf16.h>

#define B_ 8
#define T_ 2048
#define E_ 1024
#define H_ 128

typedef __attribute__((ext_vector_type(16))) __bf16 v16bf;
typedef __attribute__((ext_vector_type(8)))  __bf16 v8bf;
typedef __attribute__((ext_vector_type(8)))  float  v8f;

union bfu { v16bf v; v8bf h[2]; };

// LDS tile row strides (elements) chosen for 16B alignment + bank spread
#define KSTR 136   // K tile: 32 rows x 128 cols, 272B row stride
#define VSTR 40    // V tile: 128 rows x 32 cols, 80B row stride

__device__ __forceinline__ v8f wmma_bf16(v16bf a, v16bf b, v8f c) {
    // D = A(16x32 bf16) * B(32x16 bf16) + C(16x16 f32)
    return __builtin_amdgcn_wmma_f32_16x16x32_bf16(
        false, a, false, b, (short)0, c, false, false);
}

// CDNA5 async DMA: 16 bytes global -> LDS, tracked by ASYNCcnt
__device__ __forceinline__ void async_b128(unsigned lds_off, const void* g) {
    asm volatile("global_load_async_to_lds_b128 %0, %1, off"
                 : : "v"(lds_off), "v"(g) : "memory");
}

// ---------------------------------------------------------------------------
// Kernel 0: W (f32, [E,H]) -> wt (bf16, [3][H][E])  (transposed for contiguous
// B-fragment loads in the projection GEMM; 0.77 MB total, L2-resident)
// ---------------------------------------------------------------------------
__global__ void wconvert(const float* __restrict__ Wq,
                         const float* __restrict__ Wk,
                         const float* __restrict__ Wv,
                         __bf16* __restrict__ wt) {
    const int idx = blockIdx.x * blockDim.x + threadIdx.x;  // 0 .. 3*E*H-1
    const int sel = idx / (E_ * H_);
    const int rem = idx % (E_ * H_);
    const int e = rem / H_;
    const int h = rem % H_;
    const float* W = (sel == 0) ? Wq : ((sel == 1) ? Wk : Wv);
    wt[(size_t)sel * H_ * E_ + (size_t)h * E_ + e] = (__bf16)W[(size_t)e * H_ + h];
}

// Load one kk-step's fragments: A (16x32 from f32 x, converted) + 4 B tiles.
__device__ __forceinline__ void load_frags4(const float* __restrict__ xr,
                                            const __bf16* __restrict__ wrh,
                                            int kk, int o0, int kb0, int l16,
                                            v16bf& Af, bfu* ub) {
#pragma unroll
    for (int nt = 0; nt < 4; ++nt) {
        const __bf16* wp = wrh + (size_t)(nt * 16 + l16) * E_ + kk + kb0;
        ub[nt].h[0] = *(const v8bf*)wp;
        ub[nt].h[1] = *(const v8bf*)(wp + 8);
    }
    const float* xa = xr + kk + o0;
    const float* xb = xr + kk + 8 + o0;
#pragma unroll
    for (int i = 0; i < 8; ++i) {
        Af[i]     = (__bf16)xa[i];
        Af[8 + i] = (__bf16)xb[i];
    }
}

// ---------------------------------------------------------------------------
// Kernel 1: fused QKV projection.
//   x:[B*T, E] f32, wt:[3][H][E] bf16 -> q,k bf16 [B*T, H], vT bf16 [B][H][T]
// One wave = one 16(M)x64(N) tile for one of {q,k,v} (N-half nh of H=128).
// kk-level double buffering + sched barriers (~140 VGPRs, no spills): next
// step's loads stay in flight while the current step's 4 WMMAs drain.
// ---------------------------------------------------------------------------
__global__ void qkv_proj(const float* __restrict__ x,
                         const __bf16* __restrict__ wt,
                         __bf16* __restrict__ qb,
                         __bf16* __restrict__ kbm,
                         __bf16* __restrict__ vtb) {
    const int lane = threadIdx.x & 31;
    const int wid  = threadIdx.x >> 5;
    const int gw   = blockIdx.x * (blockDim.x >> 5) + wid;   // 0..6143
    const int mt   = gw / 6;                                 // M-tile 0..1023
    const int rem  = gw % 6;
    const int sel  = rem >> 1;                               // 0=q 1=k 2=v
    const int nh   = rem & 1;                                // N half (64 cols)

    const int hf  = lane >> 4;          // half-wave id
    const int l16 = lane & 15;
    const int o0  = hf ? 16 : 0;        // A-frag K select (contiguous k = 16h + s)
    const int kb0 = hf ? 16 : 0;        // B-frag K select
    const int row = mt * 16 + l16;      // A row handled by this lane

    v8f acc[4] = {};                    // 16x64 f32 C tiles

    const float*  xr  = x + (size_t)row * E_;
    const __bf16* wrh = wt + (size_t)sel * H_ * E_ + (size_t)nh * 64 * E_;

    v16bf Af0, Af1;
    bfu   ub0[4], ub1[4];

    load_frags4(xr, wrh, 0, o0, kb0, l16, Af0, ub0);

    for (int kk = 0; kk < E_; kk += 64) {
        load_frags4(xr, wrh, kk + 32, o0, kb0, l16, Af1, ub1);
        __builtin_amdgcn_sched_barrier(0);   // keep next loads above WMMAs
#pragma unroll
        for (int nt = 0; nt < 4; ++nt)
            acc[nt] = wmma_bf16(Af0, ub0[nt].v, acc[nt]);

        if (kk + 64 < E_)
            load_frags4(xr, wrh, kk + 64, o0, kb0, l16, Af0, ub0);
        __builtin_amdgcn_sched_barrier(0);
#pragma unroll
        for (int nt = 0; nt < 4; ++nt)
            acc[nt] = wmma_bf16(Af1, ub1[nt].v, acc[nt]);
    }

    // Epilogue: C layout -> memory (bf16)
    if (sel < 2) {
        __bf16* out = (sel == 0) ? qb : kbm;
#pragma unroll
        for (int nt = 0; nt < 4; ++nt)
#pragma unroll
            for (int r = 0; r < 8; ++r) {
                int m = r + hf * 8;
                out[(size_t)(mt * 16 + m) * H_ + nh * 64 + nt * 16 + l16] =
                    (__bf16)acc[nt][r];
            }
    } else {
        const int b  = (mt * 16) / T_;
        const int t0 = (mt * 16) % T_;
#pragma unroll
        for (int nt = 0; nt < 4; ++nt)
#pragma unroll
            for (int r = 0; r < 8; ++r) {
                int m = r + hf * 8;
                int h = nh * 64 + nt * 16 + l16;
                vtb[(size_t)b * H_ * T_ + (size_t)h * T_ + (t0 + m)] = (__bf16)acc[nt][r];
            }
    }
}

// Cooperative K/V tile staging via async DMA (one 32-key block -> LDS).
__device__ __forceinline__ void stage_kv(const __bf16* __restrict__ kbm,
                                         const __bf16* __restrict__ vtb,
                                         int b, int key0, int tid,
                                         __bf16* dK, __bf16* dV) {
    // K tile: 32 rows (keys) x 256B; 128 threads -> 4 x 16B each
    {
        const int r = tid >> 2, qd = tid & 3;
        const __bf16* gk = kbm + ((size_t)b * T_ + key0 + r) * H_ + qd * 32;
        unsigned lk = (unsigned)(uintptr_t)(dK + r * KSTR + qd * 32);
#pragma unroll
        for (int j = 0; j < 4; ++j) async_b128(lk + j * 16, gk + j * 8);
    }
    // V tile: 128 rows (h) x 64B; thread t stages row h = t
    {
        const __bf16* gv = vtb + (size_t)b * H_ * T_ + (size_t)tid * T_ + key0;
        unsigned lv = (unsigned)(uintptr_t)(dV + tid * VSTR);
#pragma unroll
        for (int j = 0; j < 4; ++j) async_b128(lv + j * 16, gv + j * 8);
    }
}

// ---------------------------------------------------------------------------
// Kernel 2: flash attention (causal), bf16 WMMA + f32 online softmax.
// Block = 128 threads (4 waves) = 64 query rows of one batch.
// K/V tiles double-buffered in LDS via async DMA; each wave owns 16 q rows
// with the full H=128 f32 O accumulator.
// ---------------------------------------------------------------------------
__global__ void attn(const __bf16* __restrict__ qb,
                     const __bf16* __restrict__ kbm,
                     const __bf16* __restrict__ vtb,
                     float* __restrict__ out) {
    __shared__ __bf16 ldsK[2][32 * KSTR];    // 2 x 8.5 KB
    __shared__ __bf16 ldsV[2][128 * VSTR];   // 2 x 10 KB
    __shared__ __bf16 ldsP[4][16 * 32];      // per-wave P staging (C->A)

    const int tid  = threadIdx.x;
    const int lane = tid & 31;
    const int wid  = tid >> 5;
    const int b    = blockIdx.x >> 5;     // batch
    const int qt   = blockIdx.x & 31;     // 64-row query tile
    const int hf   = lane >> 4;
    const int l16  = lane & 15;
    const int o0   = hf ? 16 : 0;
    const int kb0  = hf ? 16 : 0;

    const int qbase = qt * 64 + wid * 16;
    const int qrow  = qbase + l16;
    const int qhi   = qbase + 15;
    const float scale = 0.08838834764831843f;   // 1/sqrt(128)

    // Persistent Q fragments: 4 K-chunks of 32 (H = 128)
    v16bf Qf[4];
    {
        const __bf16* qp = qb + ((size_t)b * T_ + qrow) * H_;
#pragma unroll
        for (int c = 0; c < 4; ++c) {
            bfu u;
            u.h[0] = *(const v8bf*)(qp + c * 32 + o0);
            u.h[1] = *(const v8bf*)(qp + c * 32 + 8 + o0);
            Qf[c] = u.v;
        }
    }

    v8f O[8] = {};
    float mrow[8], lrow[8];
#pragma unroll
    for (int r = 0; r < 8; ++r) { mrow[r] = -1.0e30f; lrow[r] = 0.0f; }

    const int nblk = 2 * qt + 2;          // uniform per block (causal bound)

    stage_kv(kbm, vtb, b, 0, tid, &ldsK[0][0], &ldsV[0][0]);

    for (int jb = 0; jb < nblk; ++jb) {
        const int buf  = jb & 1;
        const int key0 = jb * 32;

        asm volatile("s_wait_asynccnt 0x0" ::: "memory");
        __syncthreads();                  // tile visible; prev buffer free

        if (jb + 1 < nblk)
            stage_kv(kbm, vtb, b, key0 + 32, tid, &ldsK[buf ^ 1][0], &ldsV[buf ^ 1][0]);
        if (jb + 2 < nblk)
            __builtin_prefetch(kbm + ((size_t)b * T_ + key0 + 64 + (tid >> 2)) * H_, 0, 1);

        if (key0 > qhi) continue;         // causally dead for this wave

        const __bf16* dK = &ldsK[buf][0];
        const __bf16* dV = &ldsV[buf][0];

        // ---- S = Q * K^T: batch all 8 K-fragment ds_loads, then 8 WMMAs ----
        bfu uk[8];
#pragma unroll
        for (int c = 0; c < 4; ++c) {
            const __bf16* kp0 = dK + (l16)      * KSTR + c * 32 + kb0;
            const __bf16* kp1 = dK + (16 + l16) * KSTR + c * 32 + kb0;
            uk[c].h[0]     = *(const v8bf*)kp0;
            uk[c].h[1]     = *(const v8bf*)(kp0 + 8);
            uk[4 + c].h[0] = *(const v8bf*)kp1;
            uk[4 + c].h[1] = *(const v8bf*)(kp1 + 8);
        }
        v8f c0 = {}, c1 = {};
#pragma unroll
        for (int c = 0; c < 4; ++c) {
            c0 = wmma_bf16(Qf[c], uk[c].v,     c0);
            c1 = wmma_bf16(Qf[c], uk[4 + c].v, c1);
        }

        // ---- scale + causal mask ----
        float s0[8], s1[8];
        const int keyA = key0 + l16, keyB = key0 + 16 + l16;
#pragma unroll
        for (int r = 0; r < 8; ++r) {
            const int qr = qbase + hf * 8 + r;
            float a = c0[r] * scale;
            float g = c1[r] * scale;
            if (keyA > qr) a = -1.0e30f;
            if (keyB > qr) g = -1.0e30f;
            s0[r] = a; s1[r] = g;
        }

        // ---- online softmax ----
        float alpha[8], mnew[8];
#pragma unroll
        for (int r = 0; r < 8; ++r) {
            float t = fmaxf(s0[r], s1[r]);
            t = fmaxf(t, __shfl_xor(t, 1));
            t = fmaxf(t, __shfl_xor(t, 2));
            t = fmaxf(t, __shfl_xor(t, 4));
            t = fmaxf(t, __shfl_xor(t, 8));
            const float mn = fmaxf(mrow[r], t);
            alpha[r] = __expf(mrow[r] - mn);
            mnew[r]  = mn;
            mrow[r]  = mn;
            lrow[r] *= alpha[r];
        }
#pragma unroll
        for (int nt = 0; nt < 8; ++nt)
#pragma unroll
            for (int r = 0; r < 8; ++r) O[nt][r] *= alpha[r];

        float p0[8], p1[8];
#pragma unroll
        for (int r = 0; r < 8; ++r) {
            p0[r] = __expf(s0[r] - mnew[r]);
            p1[r] = __expf(s1[r] - mnew[r]);
            float t = p0[r] + p1[r];
            t += __shfl_xor(t, 1);
            t += __shfl_xor(t, 2);
            t += __shfl_xor(t, 4);
            t += __shfl_xor(t, 8);
            lrow[r] += t;
        }

        // ---- P: C-layout -> A-layout via per-wave LDS ----
        __bf16* pl = &ldsP[wid][0];
#pragma unroll
        for (int r = 0; r < 8; ++r) {
            const int m = hf * 8 + r;
            pl[m * 32 + l16]      = (__bf16)p0[r];
            pl[m * 32 + 16 + l16] = (__bf16)p1[r];
        }
        asm volatile("s_wait_dscnt 0x0" ::: "memory");   // split DS counter
        bfu up;
        up.h[0] = *(const v8bf*)(pl + l16 * 32 + o0);
        up.h[1] = *(const v8bf*)(pl + l16 * 32 + 8 + o0);
        const v16bf Pf = up.v;

        // ---- O += P * V: batch all 8 V-fragment ds_loads, then 8 WMMAs ----
        bfu uv[8];
#pragma unroll
        for (int nt = 0; nt < 8; ++nt) {
            const __bf16* vp = dV + (nt * 16 + l16) * VSTR + kb0;
            uv[nt].h[0] = *(const v8bf*)vp;
            uv[nt].h[1] = *(const v8bf*)(vp + 8);
        }
#pragma unroll
        for (int nt = 0; nt < 8; ++nt)
            O[nt] = wmma_bf16(Pf, uv[nt].v, O[nt]);
    }

    // ---- epilogue: O / l -> out (f32) ----
#pragma unroll
    for (int r = 0; r < 8; ++r) lrow[r] = 1.0f / lrow[r];
#pragma unroll
    for (int nt = 0; nt < 8; ++nt)
#pragma unroll
        for (int r = 0; r < 8; ++r) {
            const int t = qbase + hf * 8 + r;
            out[((size_t)b * T_ + t) * H_ + nt * 16 + l16] = O[nt][r] * lrow[r];
        }
}

extern "C" void kernel_launch(void* const* d_in, const int* in_sizes, int n_in,
                              void* d_out, int out_size, void* d_ws, size_t ws_size,
                              hipStream_t stream) {
    (void)in_sizes; (void)n_in; (void)out_size; (void)ws_size;
    const float* x  = (const float*)d_in[0];
    const float* Wq = (const float*)d_in[1];
    const float* Wk = (const float*)d_in[2];
    const float* Wv = (const float*)d_in[3];
    float* out = (float*)d_out;

    const size_t N = (size_t)B_ * T_ * H_;         // 2,097,152 elements
    __bf16* qb = (__bf16*)d_ws;                    // [B*T, H]  bf16
    __bf16* kb = qb + N;                           // [B*T, H]  bf16
    __bf16* vt = kb + N;                           // [B, H, T] bf16 (transposed)
    __bf16* wt = vt + N;                           // [3][H][E] bf16 (transposed)

    wconvert<<<(3 * E_ * H_) / 256, 256, 0, stream>>>(Wq, Wk, Wv, wt);
    // 1024 M-tiles * 3 matrices * 2 N-halves = 6144 waves; 8/block -> 768
    qkv_proj<<<768, 256, 0, stream>>>(x, wt, qb, kb, vt);
    // 8 batches * 32 query tiles of 64 rows; 4 waves/block
    attn<<<256, 128, 0, stream>>>(qb, kb, vt, out);
}
